// SineRNN_44727789420874
// MI455X (gfx1250) — matrix operand order repaired and checked by reading, not compiled, not run on hardware
//
#include <hip/hip_runtime.h>
#include <hip/hip_bf16.h>

typedef __attribute__((ext_vector_type(16))) _Float16 v16h;
typedef __attribute__((ext_vector_type(8)))  float    v8f;

#define HIDDEN          32
#define ROWS_PER_WAVE   16
#define WAVES_PER_BLOCK 8
#define ROWS_PER_BLOCK  128   // 8 waves * 16 rows
#define TCHUNK          32
#define XPITCH          33    // padded pitch (floats) -> conflict-free broadcast reads

__device__ __forceinline__ float fast_tanhf(float x) {
#if __has_builtin(__builtin_amdgcn_tanhf)
  return __builtin_amdgcn_tanhf(x);          // v_tanh_f32 (CDNA5 trans op)
#else
  // tanh(x) = 1 - 2/(exp2(2*log2e*x)+1)
  float e = __builtin_amdgcn_exp2f(x * 2.8853900817779268f);
  return 1.0f - 2.0f * __builtin_amdgcn_rcpf(e + 1.0f);
#endif
}

__device__ __forceinline__ unsigned pk16(float a, float b) {
  // v_cvt_pkrtz_f16_f32 packs (a,b) -> {lo,hi} f16; bit-cast the 4-byte vector.
  return __builtin_bit_cast(unsigned, __builtin_amdgcn_cvt_pkrtz(a, b));
}

__global__ __launch_bounds__(256) void SineRNN_kernel(
    const float* __restrict__ x,     // [B, T, 1]
    const float* __restrict__ W_ih,  // [32, 1]
    const float* __restrict__ W_hh,  // [32, 32]
    const float* __restrict__ b_ih,  // [32]
    const float* __restrict__ b_hh,  // [32]
    const float* __restrict__ fc_w,  // [1, 32]
    const float* __restrict__ fc_b,  // [1]
    float* __restrict__ out,         // [B, 1]
    int T)
{
  __shared__ float xbuf[ROWS_PER_BLOCK * XPITCH];

  const int tid    = threadIdx.x;
  const int lane   = tid & 31;
  const int wv     = tid >> 5;
  const int lmod   = lane & 15;
  const bool hi    = (lane >= 16);
  const int blockb = blockIdx.x * ROWS_PER_BLOCK;
  const int waveb  = blockb + wv * ROWS_PER_WAVE;

  // ---- A matrices: W_hh tiles as WMMA A operands, this code's contiguous K order (loop invariant) ----
  // lane<16: M=lane, halves 0..15 -> K=0..15
  // lane>=16: M=lane-16, halves 0..15 -> K=16..31
  v16h a0, a1;
  {
    const int kb = hi ? 16 : 0;
#pragma unroll
    for (int i = 0; i < 8; ++i) {
      a0[i]     = (_Float16)W_hh[lmod * HIDDEN + kb + i];
      a0[8 + i] = (_Float16)W_hh[lmod * HIDDEN + kb + 8 + i];
      a1[i]     = (_Float16)W_hh[(lmod + 16) * HIDDEN + kb + i];
      a1[8 + i] = (_Float16)W_hh[(lmod + 16) * HIDDEN + kb + 8 + i];
    }
  }

  // ---- per-lane replicated vectors in C/D layout: vgpr v -> j = v + (hi?8:0) (+16 for tile1)
  float wih0[8], wih1[8], bs0[8], bs1[8], fw0[8], fw1[8];
  {
    const int jb = hi ? 8 : 0;
#pragma unroll
    for (int v = 0; v < 8; ++v) {
      const int j0 = jb + v, j1 = jb + v + 16;
      wih0[v] = W_ih[j0];
      wih1[v] = W_ih[j1];
      bs0[v]  = b_ih[j0] + b_hh[j0];
      bs1[v]  = b_ih[j1] + b_hh[j1];
      fw0[v]  = fc_w[j0];
      fw1[v]  = fc_w[j1];
    }
  }

  // ---- B matrix = h^T (32x16 f16), starts at zero (h0 = 0) ----
  union { v16h h; unsigned u[8]; } bm;
#pragma unroll
  for (int i = 0; i < 8; ++i) bm.u[i] = 0u;

  v8f d0 = {}, d1 = {};

  for (int t0 = 0; t0 < T; t0 += TCHUNK) {
    __syncthreads();
    // stage x chunk: wave loads one row's 32 contiguous floats per iter (coalesced 128B)
#pragma unroll
    for (int i = 0; i < ROWS_PER_WAVE; ++i) {
      const int row = wv * ROWS_PER_WAVE + i;
      xbuf[row * XPITCH + lane] =
          x[(size_t)(blockb + row) * (size_t)T + (size_t)(t0 + lane)];
    }
    __syncthreads();

#pragma unroll 4
    for (int tc = 0; tc < TCHUNK; ++tc) {
      // C = x_t * W_ih + (b_ih + b_hh), in D'-layout: lane's batch col = lmod
      const float xv = xbuf[(wv * ROWS_PER_WAVE + lmod) * XPITCH + tc];
      v8f c0, c1;
#pragma unroll
      for (int v = 0; v < 8; ++v) {
        c0[v] = fmaf(xv, wih0[v], bs0[v]);
        c1[v] = fmaf(xv, wih1[v], bs1[v]);
      }

      // D'[j, b] = sum_k W_hh[j,k] * h[b,k] + C  (two j-tiles)
      d0 = __builtin_amdgcn_wmma_f32_16x16x32_f16(false, a0, false, bm.h,
                                                  (short)0, c0, false, false);
      d1 = __builtin_amdgcn_wmma_f32_16x16x32_f16(false, a1, false, bm.h,
                                                  (short)0, c1, false, false);

#pragma unroll
      for (int v = 0; v < 8; ++v) {
        d0[v] = fast_tanhf(d0[v]);
        d1[v] = fast_tanhf(d1[v]);
      }

      // Repack D' (f32, M across VGPRs) into B layout (f16, K across lane halves).
      // pk0_w: low lanes j=(2w,2w+1), high lanes j=(2w+8,2w+9); pk1_w: +16.
      unsigned pk0[4], pk1[4], sw0[4], sw1[4];
#pragma unroll
      for (int w2 = 0; w2 < 4; ++w2) {
        pk0[w2] = pk16(d0[2 * w2], d0[2 * w2 + 1]);
        pk1[w2] = pk16(d1[2 * w2], d1[2 * w2 + 1]);
      }
#pragma unroll
      for (int w2 = 0; w2 < 4; ++w2) {
        sw0[w2] = (unsigned)__shfl_xor((int)pk0[w2], 16, 32);
        sw1[w2] = (unsigned)__shfl_xor((int)pk1[w2], 16, 32);
      }
      // B vgpr v: low lanes need K=2v,2v+1 ; high lanes need K=16+2v,17+2v
#pragma unroll
      for (int w2 = 0; w2 < 4; ++w2) {
        bm.u[w2]     = hi ? sw1[w2] : pk0[w2];  // K 0..7  / 16..23
        bm.u[w2 + 4] = hi ? pk1[w2] : sw0[w2];  // K 8..15 / 24..31
      }
    }
  }

  // ---- head: out[b] = sum_j h[j,b]*fc_w[j] + fc_b ----
  float part = 0.f;
#pragma unroll
  for (int v = 0; v < 8; ++v) {
    part = fmaf(d0[v], fw0[v], part);
    part = fmaf(d1[v], fw1[v], part);
  }
  part += __shfl_xor(part, 16, 32);  // combine the two j half-sets per batch col
  if (!hi) out[waveb + lmod] = part + fc_b[0];
}

extern "C" void kernel_launch(void* const* d_in, const int* in_sizes, int n_in,
                              void* d_out, int out_size, void* d_ws, size_t ws_size,
                              hipStream_t stream) {
  (void)n_in; (void)d_ws; (void)ws_size;
  const float* x    = (const float*)d_in[0];
  const float* W_ih = (const float*)d_in[1];
  const float* W_hh = (const float*)d_in[2];
  const float* b_ih = (const float*)d_in[3];
  const float* b_hh = (const float*)d_in[4];
  const float* fc_w = (const float*)d_in[5];
  const float* fc_b = (const float*)d_in[6];
  float* out = (float*)d_out;

  const int B = out_size;            // 16384
  const int T = in_sizes[0] / B;     // 1024

  const int grid = B / ROWS_PER_BLOCK;  // 128 blocks * 256 threads (8 waves)
  SineRNN_kernel<<<grid, 256, 0, stream>>>(x, W_ih, W_hh, b_ih, b_hh,
                                           fc_w, fc_b, out, T);
}
